// VQVAE_249108103721
// MI455X (gfx1250) — matrix operand order repaired and checked by reading, not compiled, not run on hardware
//
#include <hip/hip_runtime.h>
#include <hip/hip_bf16.h>

typedef __bf16 bf16;
typedef bf16     v16bf __attribute__((ext_vector_type(16)));
typedef float    v8f   __attribute__((ext_vector_type(8)));
typedef unsigned v8u   __attribute__((ext_vector_type(8)));

__device__ __forceinline__ unsigned pack_bf(float a, float b) {
  union { unsigned u; bf16 h[2]; } u;
  u.h[0] = (bf16)a; u.h[1] = (bf16)b;
  return u.u;
}

__device__ __forceinline__ v16bf frag_from(uint4 a0, uint4 a1) {
  v8u u;
  u[0] = a0.x; u[1] = a0.y; u[2] = a0.z; u[3] = a0.w;
  u[4] = a1.x; u[5] = a1.y; u[6] = a1.z; u[7] = a1.w;
  return __builtin_bit_cast(v16bf, u);
}

// ---------------------------------------------------------------------------
// Fused GroupNorm (+ optional SiLU).  One block per (batch, group), NC=32.
// Group payload is always a multiple of 1024 floats -> float4 path is safe.
// ---------------------------------------------------------------------------
__global__ __launch_bounds__(256)
void gn_silu_k(const float* __restrict__ x, float* __restrict__ y,
               const float* __restrict__ sc, const float* __restrict__ bi,
               int C, int HW, int doSilu) {
  int bg = blockIdx.x;
  int g  = bg & 31;
  int b  = bg >> 5;
  int cpg = C >> 5;
  long n  = (long)cpg * HW;
  size_t base = ((size_t)b * C + (size_t)g * cpg) * HW;
  const float4* x4 = (const float4*)(x + base);
  float4* y4 = (float4*)(y + base);
  long n4 = n >> 2;

  float s = 0.f, s2 = 0.f;
  for (long i = threadIdx.x; i < n4; i += 256) {
    float4 v = x4[i];
    s  += v.x + v.y + v.z + v.w;
    s2 += v.x * v.x + v.y * v.y + v.z * v.z + v.w * v.w;
  }
  __shared__ float r1[256], r2[256];
  r1[threadIdx.x] = s; r2[threadIdx.x] = s2;
  __syncthreads();
  for (int off = 128; off > 0; off >>= 1) {
    if ((int)threadIdx.x < off) {
      r1[threadIdx.x] += r1[threadIdx.x + off];
      r2[threadIdx.x] += r2[threadIdx.x + off];
    }
    __syncthreads();
  }
  float inv_n = 1.f / (float)n;
  float mu  = r1[0] * inv_n;
  float var = r2[0] * inv_n - mu * mu;
  float rs  = rsqrtf(var + 1e-5f);

  unsigned HW4 = (unsigned)(HW >> 2);
  for (long i = threadIdx.x; i < n4; i += 256) {
    int c = g * cpg + (int)((unsigned)i / HW4);
    float a = rs * sc[c], bb = bi[c] - mu * a;
    float4 v = x4[i];
    v.x = v.x * a + bb; v.y = v.y * a + bb; v.z = v.z * a + bb; v.w = v.w * a + bb;
    if (doSilu) {
      v.x = v.x / (1.f + __expf(-v.x));
      v.y = v.y / (1.f + __expf(-v.y));
      v.z = v.z / (1.f + __expf(-v.z));
      v.w = v.w / (1.f + __expf(-v.w));
    }
    y4[i] = v;
  }
}

// ---------------------------------------------------------------------------
// Implicit-GEMM conv via WMMA bf16.
//   mode 0: conv, weights OIHW; mode 1: ConvTranspose2d(k=4,s=2,p=1), [in,out,4,4]
// Block: 256 threads (8 waves), tile 64(M) x 128(N), K step 32.
// Tiles stored pair-packed AND fragment-contiguous:
//   AsU[m][kpair], BsU[n][kpair] with 80B rows -> every fragment = 2x ds_load_b128.
// Loader threads own consecutive k, so (ci,kh,kw) advances by carry-increment.
// ---------------------------------------------------------------------------
__global__ __launch_bounds__(256)
void conv_gemm_k(const float* __restrict__ in, const float* __restrict__ wgt,
                 const float* __restrict__ bias, const float* __restrict__ add,
                 float* __restrict__ out,
                 int Bn, int Cin, int H, int W, int Cout, int Ho, int Wo,
                 int KH, int KW, int stride, int pad, int K, int mode) {
  __shared__ __align__(16) unsigned AsU[64][20];    // [m][kpair 0..15]
  __shared__ __align__(16) unsigned BsU[128][20];   // [n][kpair 0..15]

  const int tid  = threadIdx.x;
  const int lane = tid & 31;
  const int col  = lane & 15;
  const int hi   = lane >> 4;
  const int wv   = tid >> 5;
  const int waveM = wv >> 1;
  const int waveN = wv & 1;

  const unsigned HoWo = (unsigned)(Ho * Wo);
  const unsigned Wou  = (unsigned)Wo;
  const unsigned KHWu = (unsigned)(KH * KW);
  const unsigned KWu  = (unsigned)KW;
  const unsigned KHu  = (unsigned)KH;
  const unsigned Ku   = (unsigned)K;
  const long   N    = (long)Bn * HoWo;
  const size_t HWs  = (size_t)H * W;

  const int  m0 = blockIdx.x * 64;
  const long n0 = (long)blockIdx.y * 128;

  // A loader: one row, 4 contiguous k-pairs (8 consecutive k)
  const int mmA  = tid >> 2;
  const int kpA0 = (tid & 3) * 4;
  // B loader: one column, 8 contiguous k-pairs (16 consecutive k)
  const unsigned nnB = tid & 127;
  const int kpB0 = (tid >> 7) * 8;

  const long ngB = n0 + nnB;
  const bool nOk = (ngB < N);
  int ohs = 0, ows = 0;
  unsigned ohB = 0, owB = 0;
  size_t inBaseB = 0;
  if (nOk) {
    unsigned bB  = (unsigned)((unsigned long)ngB / HoWo);
    unsigned ohw = (unsigned)ngB - bB * HoWo;
    ohB = ohw / Wou;
    owB = ohw - ohB * Wou;
    ohs = (int)(ohB * (unsigned)stride) - pad;
    ows = (int)(owB * (unsigned)stride) - pad;
    inBaseB = (size_t)bB * Cin * HWs;
  }

  v8f acc[4];
#pragma unroll
  for (int t = 0; t < 4; ++t)
#pragma unroll
    for (int i = 0; i < 8; ++i) acc[t][i] = 0.f;

  for (int k0 = 0; k0 < K; k0 += 32) {
    // ---- stage A (weights): 1 row x 8 consecutive k per thread ----
    {
      const int mg = m0 + mmA;
      const unsigned kg0 = (unsigned)k0 + 2u * kpA0;
      float v[8];
#pragma unroll
      for (int i = 0; i < 8; ++i) v[i] = 0.f;
      if (mg < Cout) {
        if (mode == 0) {
          const float* wrow = wgt + (size_t)mg * Ku + kg0;
#pragma unroll
          for (int i = 0; i < 8; ++i)
            if (kg0 + i < Ku) v[i] = wrow[i];
        } else {
          unsigned ci = kg0 / KHWu;
          unsigned r  = kg0 - ci * KHWu;
          unsigned kh = r / KWu, kw = r - kh * KWu;
#pragma unroll
          for (int i = 0; i < 8; ++i) {
            if (kg0 + i < Ku)
              v[i] = wgt[(((size_t)ci * Cout + mg) * KHu + kh) * KWu + kw];
            if (++kw == KWu) { kw = 0; if (++kh == KHu) { kh = 0; ++ci; } }
          }
        }
      }
      uint4 p0 = { pack_bf(v[0], v[1]), pack_bf(v[2], v[3]),
                   pack_bf(v[4], v[5]), pack_bf(v[6], v[7]) };
      *(uint4*)&AsU[mmA][kpA0] = p0;
    }
    // ---- stage B (im2col): 1 col x 16 consecutive k per thread ----
    {
      const unsigned kg0 = (unsigned)k0 + 2u * kpB0;
      float v[16];
#pragma unroll
      for (int i = 0; i < 16; ++i) v[i] = 0.f;
      if (nOk) {
        unsigned ci, kh, kw;
        if (KHWu == 1) { ci = kg0; kh = 0; kw = 0; }
        else {
          ci = kg0 / KHWu; unsigned r = kg0 - ci * KHWu;
          kh = r / KWu;    kw = r - kh * KWu;
        }
#pragma unroll
        for (int i = 0; i < 16; ++i) {
          if (kg0 + i < Ku) {
            int ih, iw; bool ok;
            if (mode == 0) {
              ih = ohs + (int)kh; iw = ows + (int)kw;
              ok = (ih >= 0 && ih < H && iw >= 0 && iw < W);
            } else {
              int th = (int)(ohB + 1u) - (int)kh;
              int tw = (int)(owB + 1u) - (int)kw;
              ok = (th >= 0 && tw >= 0 && !(th & 1) && !(tw & 1));
              ih = th >> 1; iw = tw >> 1;
              ok = ok && (ih < H) && (iw < W);
            }
            if (ok) v[i] = in[inBaseB + (size_t)ci * HWs + (size_t)ih * W + iw];
          }
          if (KHWu == 1) ++ci;
          else if (++kw == KWu) { kw = 0; if (++kh == KHu) { kh = 0; ++ci; } }
        }
      }
      uint4 p0 = { pack_bf(v[0],  v[1]),  pack_bf(v[2],  v[3]),
                   pack_bf(v[4],  v[5]),  pack_bf(v[6],  v[7]) };
      uint4 p1 = { pack_bf(v[8],  v[9]),  pack_bf(v[10], v[11]),
                   pack_bf(v[12], v[13]), pack_bf(v[14], v[15]) };
      *(uint4*)&BsU[nnB][kpB0]     = p0;
      *(uint4*)&BsU[nnB][kpB0 + 4] = p1;
    }
    __syncthreads();

    // ---- fragments: 2x b128 each, contiguous VGPR quads ----
    const uint4* arow = (const uint4*)&AsU[waveM * 16 + col][0];
    v16bf a = frag_from(arow[hi], arow[2 + hi]);
#pragma unroll
    for (int t = 0; t < 4; ++t) {
      const uint4* brow = (const uint4*)&BsU[waveN * 64 + t * 16 + col][0];
      v16bf bv = frag_from(brow[hi], brow[2 + hi]);
      acc[t] = __builtin_amdgcn_wmma_f32_16x16x32_bf16(false, a, false, bv, (short)0, acc[t], false, false);
    }
    __syncthreads();
  }

  // ---- epilogue (C/D layout: n=lane&15, rows m=(lane>>4)*8+i) ----
  const int mBase = m0 + waveM * 16 + hi * 8;
#pragma unroll
  for (int t = 0; t < 4; ++t) {
    long n = n0 + waveN * 64 + t * 16 + col;
    if (n < N) {
      unsigned b   = (unsigned)((unsigned long)n / HoWo);
      unsigned ohw = (unsigned)n - b * HoWo;
      size_t idxBase = (size_t)b * Cout * HoWo + ohw;
#pragma unroll
      for (int i = 0; i < 8; ++i) {
        int m = mBase + i;
        if (m < Cout) {
          size_t idx = idxBase + (size_t)m * HoWo;
          float v = acc[t][i] + bias[m];
          if (add) v += add[idx];
          out[idx] = v;
        }
      }
    }
  }
}

// ---------------------------------------------------------------------------
// Flash-style attention core. qkv: (B, 768, L) f32, channels [q|k|v] of 256.
// NH=4, hd=64. Grid: B*NH*(L/128) blocks of 256 threads; each wave owns 16
// query rows.  All tiles stored fragment-contiguous (2x ds_load_b128 each).
// Writes O as (B, 256, L).
// ---------------------------------------------------------------------------
__global__ __launch_bounds__(256)
void attn_core_k(const float* __restrict__ qkv, float* __restrict__ o, int L) {
  const int qtiles = L / 128;
  int blk = blockIdx.x;
  int qt = blk % qtiles;
  int bh = blk / qtiles;
  int h = bh & 3;
  int b = bh >> 2;

  const int tid  = threadIdx.x;
  const int w    = tid >> 5;
  const int lane = tid & 31;
  const int col  = lane & 15;
  const int hi   = lane >> 4;
  const int qBase = qt * 128 + w * 16;

  const size_t baseQ = ((size_t)b * 768 + h * 64) * L;
  const size_t baseK = baseQ + (size_t)256 * L;
  const size_t baseV = baseQ + (size_t)512 * L;

  __shared__ __align__(16) unsigned QsU[8][16][36];  // [wave][row][dpair]
  __shared__ __align__(16) unsigned KsT[32][36];     // [key][dpair]
  __shared__ __align__(16) unsigned VsT[64][20];     // [d][kpair]
  __shared__ __align__(16) bf16     Ps[8][16][40];   // [wave][row][key]

  // Q tile: row r, d-pair = lane (covers d 0..63)
  for (int r = 0; r < 16; ++r) {
    float q0 = qkv[baseQ + (size_t)(2 * lane) * L + (qBase + r)];
    float q1 = qkv[baseQ + (size_t)(2 * lane + 1) * L + (qBase + r)];
    QsU[w][r][lane] = pack_bf(q0, q1);
  }

  float rm[8], rl[8];
  v8f acc[4];
#pragma unroll
  for (int i = 0; i < 8; ++i) { rm[i] = -1e30f; rl[i] = 0.f; }
#pragma unroll
  for (int t = 0; t < 4; ++t)
#pragma unroll
    for (int i = 0; i < 8; ++i) acc[t][i] = 0.f;

  const int keyK = tid & 31;          // K stage: fixed key
  const int dpK0 = (tid >> 5) * 4;    // K stage: 4 contiguous d-pairs
  const int dV   = tid & 63;          // V stage: fixed d
  const int kpV0 = (tid >> 6) * 4;    // V stage: 4 contiguous k-pairs

  for (int kb = 0; kb < L / 32; ++kb) {
    __syncthreads();
    {
      float kv[8];
#pragma unroll
      for (int i = 0; i < 8; ++i)
        kv[i] = qkv[baseK + (size_t)(2 * dpK0 + i) * L + (kb * 32 + keyK)];
      uint4 pk = { pack_bf(kv[0], kv[1]), pack_bf(kv[2], kv[3]),
                   pack_bf(kv[4], kv[5]), pack_bf(kv[6], kv[7]) };
      *(uint4*)&KsT[keyK][dpK0] = pk;
    }
    {
      const float4* vp = (const float4*)&qkv[baseV + (size_t)dV * L + (kb * 32 + 2 * kpV0)];
      float4 va = vp[0], vb = vp[1];
      uint4 pk = { pack_bf(va.x, va.y), pack_bf(va.z, va.w),
                   pack_bf(vb.x, vb.y), pack_bf(vb.z, vb.w) };
      *(uint4*)&VsT[dV][kpV0] = pk;
    }
    __syncthreads();

    // S = Q @ K^T : two 16-key n-tiles, d=64 contraction (2 WMMA steps each)
    v8f s[2];
    const uint4* qrow = (const uint4*)&QsU[w][col][0];
#pragma unroll
    for (int nt = 0; nt < 2; ++nt) {
      v8f z;
#pragma unroll
      for (int i = 0; i < 8; ++i) z[i] = 0.f;
      const uint4* krow = (const uint4*)&KsT[nt * 16 + col][0];
#pragma unroll
      for (int ks = 0; ks < 2; ++ks) {
        v16bf a  = frag_from(qrow[ks * 4 + hi], qrow[ks * 4 + 2 + hi]);
        v16bf bk = frag_from(krow[ks * 4 + hi], krow[ks * 4 + 2 + hi]);
        z = __builtin_amdgcn_wmma_f32_16x16x32_bf16(false, a, false, bk, (short)0, z, false, false);
      }
      s[nt] = z;
    }

    // online softmax per row (rows hi*8+i live in the same 16-lane half)
    const float scl = 0.125f;  // 1/sqrt(64)
#pragma unroll
    for (int i = 0; i < 8; ++i) {
      float s0 = s[0][i] * scl, s1 = s[1][i] * scl;
      float mx = fmaxf(s0, s1);
      for (int off = 1; off < 16; off <<= 1) mx = fmaxf(mx, __shfl_xor(mx, off, 32));
      float nm  = fmaxf(rm[i], mx);
      float cor = __expf(rm[i] - nm);
      float p0  = __expf(s0 - nm);
      float p1  = __expf(s1 - nm);
      float rs  = p0 + p1;
      for (int off = 1; off < 16; off <<= 1) rs += __shfl_xor(rs, off, 32);
      rl[i] = rl[i] * cor + rs;
      rm[i] = nm;
#pragma unroll
      for (int t = 0; t < 4; ++t) acc[t][i] *= cor;
      int row = hi * 8 + i;
      Ps[w][row][col]      = (bf16)p0;
      Ps[w][row][16 + col] = (bf16)p1;
    }
    __syncthreads();

    // O += P @ V  (K=32 keys, 4 dim n-tiles)
    const uint4* prow = (const uint4*)&Ps[w][col][0];
    v16bf aP = frag_from(prow[hi], prow[2 + hi]);
#pragma unroll
    for (int dt = 0; dt < 4; ++dt) {
      const uint4* vrow = (const uint4*)&VsT[dt * 16 + col][0];
      v16bf bv = frag_from(vrow[hi], vrow[2 + hi]);
      acc[dt] = __builtin_amdgcn_wmma_f32_16x16x32_bf16(false, aP, false, bv, (short)0, acc[dt], false, false);
    }
  }

  const size_t baseO = ((size_t)b * 256 + h * 64) * L;
#pragma unroll
  for (int dt = 0; dt < 4; ++dt)
#pragma unroll
    for (int i = 0; i < 8; ++i) {
      int row = hi * 8 + i;
      o[baseO + (size_t)(dt * 16 + col) * L + (qBase + row)] = acc[dt][i] / rl[i];
    }
}

// ---------------------------------------------------------------------------
// Nearest-codebook quantization. z: (16,4,32,32) NCHW. Codebook 8192x4.
// ---------------------------------------------------------------------------
__global__ __launch_bounds__(256)
void quantize_k(const float* __restrict__ z, const float* __restrict__ E,
                float* __restrict__ zq, float* __restrict__ zq2,
                float* __restrict__ lossAcc, int L) {
  __shared__ float cb[4096 * 4];   // 64 KB chunk
  int p = blockIdx.x * 256 + threadIdx.x;
  int b = p / L, l = p - b * L;
  size_t base = (size_t)b * 4 * L + l;
  float z0 = z[base], z1 = z[base + L], z2 = z[base + 2 * L], z3 = z[base + 3 * L];
  int best = 0; float bd = 3.4e38f;

  for (int ch = 0; ch < 2; ++ch) {
    __syncthreads();
    for (int t = threadIdx.x; t < 4096 * 4; t += 256)
      cb[t] = E[ch * 4096 * 4 + t];
    __syncthreads();
    for (int k = 0; k < 4096; ++k) {
      float d0 = z0 - cb[4 * k], d1 = z1 - cb[4 * k + 1];
      float d2 = z2 - cb[4 * k + 2], d3 = z3 - cb[4 * k + 3];
      float d = d0 * d0 + d1 * d1 + d2 * d2 + d3 * d3;
      if (d < bd) { bd = d; best = ch * 4096 + k; }
    }
  }
  float e0 = E[4 * best], e1 = E[4 * best + 1], e2 = E[4 * best + 2], e3 = E[4 * best + 3];
  zq[base] = e0;  zq[base + L] = e1;  zq[base + 2 * L] = e2;  zq[base + 3 * L] = e3;
  zq2[base] = e0; zq2[base + L] = e1; zq2[base + 2 * L] = e2; zq2[base + 3 * L] = e3;

  __shared__ float red[256];
  red[threadIdx.x] = bd;
  __syncthreads();
  for (int off = 128; off > 0; off >>= 1) {
    if ((int)threadIdx.x < off) red[threadIdx.x] += red[threadIdx.x + off];
    __syncthreads();
  }
  if (threadIdx.x == 0) atomicAdd(lossAcc, red[0]);
}

__global__ void loss_write_k(const float* __restrict__ lossAcc, float* __restrict__ out) {
  float v = lossAcc[0] / (16.f * 1024.f * 4.f);
  out[0] = v;   // codebook_loss (forward-identical to commit_loss)
  out[1] = v;   // commit_loss
}

// ---------------------------------------------------------------------------
// Host orchestration
// ---------------------------------------------------------------------------
extern "C" void kernel_launch(void* const* d_in, const int* in_sizes, int n_in,
                              void* d_out, int out_size, void* d_ws, size_t ws_size,
                              hipStream_t stream) {
  (void)in_sizes; (void)n_in; (void)out_size; (void)ws_size;
  int cur = 0;
  auto nf = [&]() -> const float* { return (const float*)d_in[cur++]; };

  const float* x = nf();                       // (16,3,128,128)
  const int Bn = 16;

  float* ws = (float*)d_ws;
  const size_t CAP = (size_t)16 * 128 * 128 * 128;   // 33,554,432 floats
  float* A  = ws;
  float* Bf = A  + CAP;
  float* Cf = Bf + CAP;
  float* Df = Cf + CAP;
  float* zbuf    = Df + CAP;                   // 65,536 floats
  float* lossAcc = zbuf + (size_t)16 * 4 * 1024;

  auto conv = [&](const float* in, const float* w, const float* bi, const float* add,
                  float* out, int Cin, int H, int W, int Cout,
                  int KH, int KW, int stride, int pad, int mode) {
    int Ho = (mode == 0) ? (H + 2 * pad - KH) / stride + 1 : 2 * H;
    int Wo = (mode == 0) ? (W + 2 * pad - KW) / stride + 1 : 2 * W;
    int K  = Cin * KH * KW;
    long N = (long)Bn * Ho * Wo;
    dim3 grid((unsigned)((Cout + 63) / 64), (unsigned)((N + 127) / 128));
    conv_gemm_k<<<grid, 256, 0, stream>>>(in, w, bi, add, out, Bn, Cin, H, W, Cout,
                                          Ho, Wo, KH, KW, stride, pad, K, mode);
  };
  auto gn = [&](const float* in, float* out, const float* s, const float* b,
                int C, int HW, int silu) {
    gn_silu_k<<<Bn * 32, 256, 0, stream>>>(in, out, s, b, C, HW, silu);
  };
  auto resnetB = [&](const float* in, float* out, float* t1, float* t2,
                     int Cin, int Cout, int H, int W) {
    const float* g1s = nf(); const float* g1b = nf();
    const float* w1  = nf(); const float* b1  = nf();
    const float* g2s = nf(); const float* g2b = nf();
    const float* w2  = nf(); const float* b2  = nf();
    const float* wr  = nf(); const float* br  = nf();
    gn(in, t1, g1s, g1b, Cin, H * W, 1);
    conv(t1, w1, b1, nullptr, t2, Cin, H, W, Cout, 3, 3, 1, 1, 0);
    gn(t2, t1, g2s, g2b, Cout, H * W, 1);
    conv(in, wr, br, nullptr, out, Cin, H, W, Cout, 1, 1, 1, 0, 0);      // skip
    conv(t1, w2, b2, out, out, Cout, H, W, Cout, 3, 3, 1, 1, 0);         // + skip
  };
  auto attnB = [&](const float* xin, float* out, float* t1, float* t2, int H, int W) {
    const float* gs = nf(); const float* gb = nf();
    const float* wqkv = nf(); const float* bq = nf();
    const float* wo = nf(); const float* bo = nf();
    int L = H * W;
    gn(xin, t1, gs, gb, 256, L, 0);
    conv(t1, wqkv, bq, nullptr, t2, 256, H, W, 768, 1, 1, 1, 0, 0);      // qkv
    attn_core_k<<<Bn * 4 * (L / 128), 256, 0, stream>>>(t2, t1, L);      // O -> t1
    conv(t1, wo, bo, xin, out, 256, H, W, 256, 1, 1, 1, 0, 0);           // proj + res
  };

  // ----------------- encoder -----------------
  { const float* w = nf(); const float* b = nf();
    conv(x, w, b, nullptr, A, 3, 128, 128, 64, 3, 3, 1, 1, 0); }          // enc_in -> A
  resnetB(A, Bf, Cf, Df, 64, 128, 128, 128);                              // down0.res -> B
  { const float* w = nf(); const float* b = nf();
    conv(Bf, w, b, nullptr, Cf, 128, 128, 128, 128, 4, 4, 2, 1, 0); }     // down0 -> C @64^2
  resnetB(Cf, A, Bf, Df, 128, 256, 64, 64);                               // down1.res -> A
  { const float* w = nf(); const float* b = nf();
    conv(A, w, b, nullptr, Bf, 256, 64, 64, 256, 4, 4, 2, 1, 0); }        // down1 -> B @32^2
  resnetB(Bf, A, Cf, Df, 256, 256, 32, 32);                               // mid.res0 -> A
  attnB(A, Bf, Cf, Df, 32, 32);                                           // mid.attn -> B
  resnetB(Bf, A, Cf, Df, 256, 256, 32, 32);                               // mid.res1 -> A
  { const float* s = nf(); const float* b = nf(); gn(A, Bf, s, b, 256, 1024, 1); }
  { const float* w = nf(); const float* b = nf();
    conv(Bf, w, b, nullptr, Cf, 256, 32, 32, 4, 3, 3, 1, 1, 0); }         // enc_out -> C
  { const float* w = nf(); const float* b = nf();
    conv(Cf, w, b, nullptr, Df, 4, 32, 32, 4, 1, 1, 1, 0, 0); }           // pre_quant -> D

  // ----------------- quantize -----------------
  const float* codebook = nf();
  hipMemsetAsync(lossAcc, 0, sizeof(float), stream);
  float* zsec = (float*)d_out + (size_t)16 * 3 * 128 * 128;
  quantize_k<<<64, 256, 0, stream>>>(Df, codebook, zbuf, zsec, lossAcc, 1024);
  loss_write_k<<<1, 1, 0, stream>>>(lossAcc, zsec + (size_t)16 * 4 * 1024);

  // ----------------- decoder -----------------
  { const float* w = nf(); const float* b = nf();
    conv(zbuf, w, b, nullptr, Cf, 4, 32, 32, 4, 1, 1, 1, 0, 0); }         // post_quant -> C
  { const float* w = nf(); const float* b = nf();
    conv(Cf, w, b, nullptr, A, 4, 32, 32, 256, 3, 3, 1, 1, 0); }          // dec_in -> A
  resnetB(A, Bf, Cf, Df, 256, 256, 32, 32);                               // mid.res0 -> B
  attnB(Bf, A, Cf, Df, 32, 32);                                           // mid.attn -> A
  resnetB(A, Bf, Cf, Df, 256, 256, 32, 32);                               // mid.res1 -> B
  { const float* w = nf(); const float* b = nf();
    conv(Bf, w, b, nullptr, A, 256, 32, 32, 256, 4, 4, 2, 1, 1); }        // up0 convT -> A @64^2
  resnetB(A, Bf, Cf, Df, 256, 128, 64, 64);                               // up0.res -> B
  { const float* w = nf(); const float* b = nf();
    conv(Bf, w, b, nullptr, A, 128, 64, 64, 128, 4, 4, 2, 1, 1); }        // up1 convT -> A @128^2
  resnetB(A, Bf, Cf, Df, 128, 64, 128, 128);                              // up1.res -> B
  { const float* s = nf(); const float* b = nf(); gn(Bf, A, s, b, 64, 16384, 1); }
  { const float* w = nf(); const float* b = nf();
    conv(A, w, b, nullptr, (float*)d_out, 64, 128, 128, 3, 3, 3, 1, 1, 0); } // dec_out
}
